// E75MultiHeadCell_11416023073343
// MI455X (gfx1250) — matrix operand order repaired and checked by reading, not compiled, not run on hardware
//
#include <hip/hip_runtime.h>
#include <hip/hip_bf16.h>
#include <math.h>

#define T_STEPS 1024
#define BATCH   32
#define DIM     1024
#define HEADS   8
#define NDIM    64
#define HN      512                    // HEADS*NDIM
#define M_TOTAL (T_STEPS * BATCH)      // 32768
#define N_TOTAL (4 * HN)               // 2048  (k,v,q,beta concatenated)

typedef __attribute__((ext_vector_type(16))) __bf16 v16bf;
typedef __attribute__((ext_vector_type(8)))  float  v8f;
typedef int v2i32 __attribute__((vector_size(8)));

#ifndef __has_builtin
#define __has_builtin(x) 0
#endif

// --- gfx1250 fast transcendentals (guarded; fall back to libm) -------------
#if __has_builtin(__builtin_amdgcn_tanhf)
#define TANHF(x) __builtin_amdgcn_tanhf(x)
#elif __has_builtin(__builtin_amdgcn_tanh_f32)
#define TANHF(x) __builtin_amdgcn_tanh_f32(x)
#else
#define TANHF(x) tanhf(x)
#endif

#if __has_builtin(__builtin_amdgcn_exp2f) && __has_builtin(__builtin_amdgcn_rcpf)
#define FAST_SIG(x) __builtin_amdgcn_rcpf(1.0f + __builtin_amdgcn_exp2f(-1.4426950408889634f * (x)))
#else
#define FAST_SIG(x) (1.0f / (1.0f + __expf(-(x))))
#endif

// --- gfx1250 async global->LDS copy (guarded; fall back to ds stores) ------
#if __has_builtin(__builtin_amdgcn_global_load_async_to_lds_b64) && \
    __has_builtin(__builtin_amdgcn_s_wait_asynccnt)
#define HAVE_ASYNC_LDS 1
#else
#define HAVE_ASYNC_LDS 0
#endif

#if HAVE_ASYNC_LDS
__device__ __forceinline__ void async_copy_b64(const float* g, float* l) {
    __builtin_amdgcn_global_load_async_to_lds_b64(
        (__attribute__((address_space(1))) v2i32*)g,
        (__attribute__((address_space(3))) v2i32*)l,
        /*imm offset*/0, /*cpol*/0);
}
#endif

__device__ __forceinline__ __bf16 f2bf(float f) { return (__bf16)f; }

// ---------------------------------------------------------------------------
// Phase 1: fused projection GEMM  C[m, n] = sum_d x[m,d] * Wcat[n,d]
// Wcat = [W_k; W_v; W_q; W_beta] (each 512x1024).  3xBF16 split for ~f32 acc.
// Block tile 128(M) x 64(N), K-step 32, 8 waves (4x2), wave tile 32x32.
// ---------------------------------------------------------------------------
__global__ __launch_bounds__(256)
void proj_gemm_wmma(const float* __restrict__ x,
                    const float* __restrict__ Wk, const float* __restrict__ Wv,
                    const float* __restrict__ Wq, const float* __restrict__ Wb,
                    float* __restrict__ proj)  // [4][M_TOTAL][HN] f32
{
    __shared__ __bf16 Ah[128][40];
    __shared__ __bf16 Al[128][40];
    __shared__ __bf16 Bh[64][40];
    __shared__ __bf16 Bl[64][40];

    const int tid = threadIdx.x;
    const int m0  = blockIdx.x * 128;
    const int n0  = blockIdx.y * 64;           // global col in [0, 2048)
    const int p   = n0 >> 9;                   // which projection matrix
    const float* __restrict__ W =
        (p == 0) ? Wk : (p == 1) ? Wv : (p == 2) ? Wq : Wb;
    const int wrow0 = n0 & 511;

    const int wave = tid >> 5;
    const int lane = tid & 31;
    const int wm   = wave & 3;                 // 4 waves along M
    const int wn   = wave >> 2;                // 2 waves along N
    const int lr   = lane & 15;
    const int half = lane >> 4;

    v8f acc[2][2] = {};

    for (int k0 = 0; k0 < DIM; k0 += 32) {
        // --- stage A tile 128x32 f32 -> bf16 hi/lo ---
        #pragma unroll
        for (int it = 0; it < 4; ++it) {
            const int idx = tid + it * 256;            // 0..1023
            const int row = idx >> 3;
            const int c4  = (idx & 7) << 2;
            const float4 v =
                *(const float4*)(x + (size_t)(m0 + row) * DIM + k0 + c4);
            const float vv[4] = {v.x, v.y, v.z, v.w};
            #pragma unroll
            for (int e = 0; e < 4; ++e) {
                const __bf16 hi = f2bf(vv[e]);
                Ah[row][c4 + e] = hi;
                Al[row][c4 + e] = f2bf(vv[e] - (float)hi);
            }
        }
        // --- stage B tile 64x32 (weight rows) ---
        #pragma unroll
        for (int it = 0; it < 2; ++it) {
            const int idx = tid + it * 256;            // 0..511
            const int row = idx >> 3;
            const int c4  = (idx & 7) << 2;
            const float4 v =
                *(const float4*)(W + (size_t)(wrow0 + row) * DIM + k0 + c4);
            const float vv[4] = {v.x, v.y, v.z, v.w};
            #pragma unroll
            for (int e = 0; e < 4; ++e) {
                const __bf16 hi = f2bf(vv[e]);
                Bh[row][c4 + e] = hi;
                Bl[row][c4 + e] = f2bf(vv[e] - (float)hi);
            }
        }
        __syncthreads();

        // --- fragments (A and B both in contiguous k = 16*half + s order) ---
        v16bf afh[2], afl[2], bfh[2], bfl[2];
        #pragma unroll
        for (int mi = 0; mi < 2; ++mi) {
            const int row = wm * 32 + mi * 16 + lr;
            const int kb  = half * 16;
            #pragma unroll
            for (int e = 0; e < 8; ++e) {
                afh[mi][e]     = Ah[row][kb + e];
                afh[mi][8 + e] = Ah[row][kb + 8 + e];
                afl[mi][e]     = Al[row][kb + e];
                afl[mi][8 + e] = Al[row][kb + 8 + e];
            }
        }
        #pragma unroll
        for (int ni = 0; ni < 2; ++ni) {
            const int row = wn * 32 + ni * 16 + lr;    // B column index
            const int kb  = half * 16;
            #pragma unroll
            for (int e = 0; e < 16; ++e) {
                bfh[ni][e] = Bh[row][kb + e];
                bfl[ni][e] = Bl[row][kb + e];
            }
        }

        // --- 3xBF16 accumulate: hi*hi + hi*lo + lo*hi ---
        #pragma unroll
        for (int mi = 0; mi < 2; ++mi)
            #pragma unroll
            for (int ni = 0; ni < 2; ++ni) {
                acc[mi][ni] = __builtin_amdgcn_wmma_f32_16x16x32_bf16(
                    false, afh[mi], false, bfh[ni], (short)0, acc[mi][ni], false, false);
                acc[mi][ni] = __builtin_amdgcn_wmma_f32_16x16x32_bf16(
                    false, afh[mi], false, bfl[ni], (short)0, acc[mi][ni], false, false);
                acc[mi][ni] = __builtin_amdgcn_wmma_f32_16x16x32_bf16(
                    false, afl[mi], false, bfh[ni], (short)0, acc[mi][ni], false, false);
            }
        __syncthreads();
    }

    // --- epilogue: C/D layout -> proj[p][m][col] ---
    const int mofs = (lane >= 16) ? 8 : 0;
    #pragma unroll
    for (int mi = 0; mi < 2; ++mi)
        #pragma unroll
        for (int ni = 0; ni < 2; ++ni)
            #pragma unroll
            for (int r = 0; r < 8; ++r) {
                const int gm  = m0 + wm * 32 + mi * 16 + r + mofs;
                const int gn  = n0 + wn * 32 + ni * 16 + lr;
                const int pp  = gn >> 9;
                const int col = gn & 511;
                proj[((size_t)pp * M_TOTAL + gm) * HN + col] = acc[mi][ni][r];
            }
}

// ---------------------------------------------------------------------------
// Phase 2: per-(t,b,h) group of 64: k <- k / (||k|| + eps),
//          beta <- sigmoid(beta + b_beta[h]).  One wave32 per group.
// ---------------------------------------------------------------------------
__global__ __launch_bounds__(256)
void norm_sigmoid(float* __restrict__ projK, float* __restrict__ projB,
                  const float* __restrict__ b_beta)
{
    const int g    = blockIdx.x * 8 + (threadIdx.x >> 5);  // [0, T*B*H)
    const int lane = threadIdx.x & 31;
    const int m    = g >> 3;
    const int h    = g & 7;

    float* kp = projK + (size_t)m * HN + h * NDIM;
    const float a = kp[lane];
    const float c = kp[lane + 32];
    float ss = fmaf(a, a, c * c);
    #pragma unroll
    for (int off = 16; off >= 1; off >>= 1) ss += __shfl_xor(ss, off, 32);
    const float inv = 1.0f / (sqrtf(ss) + 1e-6f);
    kp[lane]      = a * inv;
    kp[lane + 32] = c * inv;

    float* bp = projB + (size_t)m * HN + h * NDIM;
    const float* bb = b_beta + h * NDIM;
    const float b0 = bp[lane]      + bb[lane];
    const float b1 = bp[lane + 32] + bb[lane + 32];
    bp[lane]      = FAST_SIG(b0);
    bp[lane + 32] = FAST_SIG(b1);
}

// ---------------------------------------------------------------------------
// Phase 3: sequential scan. One block (= 2 waves) per (b,h); thread i owns
// row i of S in 64 registers.  k,q stream into double-buffered LDS via
// async global->LDS copies pipelined one step ahead; one barrier per step.
// ---------------------------------------------------------------------------
__global__ __launch_bounds__(64)
void delta_scan(const float* __restrict__ proj, float* __restrict__ out)
{
    const int bh = blockIdx.x;
    const int b  = bh >> 3;
    const int h  = bh & 7;
    const int i  = threadIdx.x;

    const size_t PS = (size_t)M_TOTAL * HN;
    const float* __restrict__ Kp = proj;
    const float* __restrict__ Vp = proj + PS;
    const float* __restrict__ Qp = proj + 2 * PS;
    const float* __restrict__ Bp = proj + 3 * PS;

    __shared__ float sk2[2][NDIM];
    __shared__ float sq2[2][NDIM];

    float S[NDIM];
    #pragma unroll
    for (int j = 0; j < NDIM; ++j) S[j] = 0.0f;

    const size_t row0   = (size_t)b * HN + h * NDIM;
    const size_t stride = (size_t)BATCH * HN;

#if HAVE_ASYNC_LDS
    {   // preload t = 0: wave 0 copies k, wave 1 copies q (b64 per lane)
        const int hf = i >> 5;
        const int e  = (i & 31) * 2;
        const float* src = (hf ? Qp : Kp) + row0 + e;
        float*       dst = hf ? &sq2[0][e] : &sk2[0][e];
        async_copy_b64(src, dst);
    }
#endif

    for (int t = 0; t < T_STEPS; ++t) {
        const int    p      = t & 1;
        const size_t rowoff = row0 + (size_t)t * stride;
        const float  vv = Vp[rowoff + i];
        const float  be = Bp[rowoff + i];

#if HAVE_ASYNC_LDS
        __builtin_amdgcn_s_wait_asynccnt(0);   // step t's k,q landed in LDS
        __syncthreads();                        // visible to both waves
        if (t + 1 < T_STEPS) {                  // pipeline t+1 into other buffer
            const int hf = i >> 5;
            const int e  = (i & 31) * 2;
            const size_t nrow = rowoff + stride;
            const float* src = (hf ? Qp : Kp) + nrow + e;
            float*       dst = hf ? &sq2[p ^ 1][e] : &sk2[p ^ 1][e];
            async_copy_b64(src, dst);
            __builtin_prefetch(Vp + nrow + i, 0, 1);
            __builtin_prefetch(Bp + nrow + i, 0, 1);
        }
#else
        sk2[p][i] = Kp[rowoff + i];
        sq2[p][i] = Qp[rowoff + i];
        __syncthreads();
#endif
        const float* sk = sk2[p];
        const float* sq = sq2[p];

        // retrieved_i = sum_j S[i][j] * k[j]
        float r = 0.0f;
        #pragma unroll
        for (int j = 0; j < NDIM; ++j) r = fmaf(S[j], sk[j], r);
        const float delta = vv - r;

        // S[i][j] = tanh(beta_i * S[i][j] + delta_i * k[j])
        #pragma unroll
        for (int j = 0; j < NDIM; ++j)
            S[j] = TANHF(fmaf(be, S[j], delta * sk[j]));

        // Sq_i = sum_j S[i][j] * q[j];  out = Sq*silu(Sq) = Sq^2 * sigmoid(Sq)
        float sqv = 0.0f;
        #pragma unroll
        for (int j = 0; j < NDIM; ++j) sqv = fmaf(S[j], sq[j], sqv);

        out[rowoff + i] = sqv * sqv * FAST_SIG(sqv);
        // no trailing barrier: double buffering + the top-of-step barrier
        // already fence buffer reuse (reads of buf p finish before a thread
        // signals the barrier that gates the next overwrite of buf p).
    }

    // S_final: [B, H, n, n] appended after the [T,B,HN] output block
    float* __restrict__ Sout = out + (size_t)T_STEPS * BATCH * HN;
    #pragma unroll
    for (int j = 0; j < NDIM; ++j)
        Sout[(((size_t)b * HEADS + h) * NDIM + i) * NDIM + j] = S[j];
}

// ---------------------------------------------------------------------------
extern "C" void kernel_launch(void* const* d_in, const int* in_sizes, int n_in,
                              void* d_out, int out_size, void* d_ws, size_t ws_size,
                              hipStream_t stream)
{
    (void)in_sizes; (void)n_in; (void)out_size; (void)ws_size;
    const float* x  = (const float*)d_in[0];
    const float* Wk = (const float*)d_in[1];
    const float* Wv = (const float*)d_in[2];
    const float* Wq = (const float*)d_in[3];
    const float* Wb = (const float*)d_in[4];
    const float* bb = (const float*)d_in[5];
    float* proj = (float*)d_ws;   // 4 * 32768 * 512 f32 = 256 MiB scratch
    float* out  = (float*)d_out;

    dim3 g1(M_TOTAL / 128, N_TOTAL / 64);   // 256 x 32 blocks
    proj_gemm_wmma<<<g1, dim3(256), 0, stream>>>(x, Wk, Wv, Wq, Wb, proj);

    const size_t PS = (size_t)M_TOTAL * HN;
    norm_sigmoid<<<dim3((T_STEPS * BATCH * HEADS) / 8), dim3(256), 0, stream>>>(
        proj, proj + 3 * PS, bb);

    delta_scan<<<dim3(BATCH * HEADS), dim3(64), 0, stream>>>(proj, out);
}
